// GNN_infer_51539607552500
// MI455X (gfx1250) — matrix operand order repaired and checked by reading, not compiled, not run on hardware
//
#include <hip/hip_runtime.h>
#include <hip/hip_bf16.h>
#include <math.h>

// ---------------------------------------------------------------------------
// CDNA5 (gfx1250) implementation. wave32.
// All channel contractions on v_wmma_f32_16x16x32_f16 (f16 ops, f32 accum).
// Activations: pixel-major (P, Cs) with Cs a multiple of 16; 10-ch groups in
// 16-ch slots. Weights pre-packed on device into WMMA A-fragment layout with
// zero padding (M and K) -> branch-free inner loops, b128 loads throughout.
// ---------------------------------------------------------------------------

typedef _Float16 half_t;
typedef __attribute__((ext_vector_type(16))) _Float16 v16h;
typedef __attribute__((ext_vector_type(8)))  float    v8f;

#define PIX   18432      // N*H*W
#define HW    9216       // H*W
#define IMW   96
#define HD_   10

// K index held in half-slot h (0..15) of the A fragment, for lane-half kh.
__device__ __forceinline__ int a_k_of(int h, int kh) {
  return (kh << 4) + h;
}
__device__ __forceinline__ float sigmoidf_(float x) {
  return 1.0f / (1.0f + __expf(-x));
}

// ---------------------------------------------------------------------------
// Weight packing into WMMA A-fragment layout (zero padded).
// Fragment layout: ((mt*ktn + kt)*32 + lane)*16 + h   (halves)
// K mapping: if kvalid==0: src=k (k<Ksrc). else: k<kbase -> src=k; otherwise
// slot s=(k-kbase)/16, w=(k-kbase)%16 -> src = kbase + s*kvalid + w (w<kvalid).
// M mapping (gemm only): if mvalid: slot rows, real m = (m/16)*mvalid + m%16.
// ---------------------------------------------------------------------------
__global__ void pack_a_gemm(const float* __restrict__ Wsrc, half_t* __restrict__ F,
                            int Mreal, int Ksrc, int ktn, int kbase, int kvalid,
                            int mvalid, int ntot) {
  int i = blockIdx.x * blockDim.x + threadIdx.x;
  if (i >= ntot) return;
  int h = i & 15, lane = (i >> 4) & 31;
  int t = i >> 9;
  int kt = t % ktn, mt = t / ktn;
  int mrow = mt * 16 + (lane & 15);
  int kh = lane >> 4;
  int k = kt * 32 + a_k_of(h, kh);
  int mr;
  if (mvalid) { int s = mrow >> 4, w = mrow & 15; mr = (w < mvalid) ? s * mvalid + w : -1; }
  else mr = mrow;
  int src;
  if (kvalid == 0) src = (k < Ksrc) ? k : -1;
  else if (k < kbase) src = k;
  else {
    int s = (k - kbase) >> 4, w = (k - kbase) & 15;
    int c2 = kbase + s * kvalid + w;
    src = (w < kvalid && c2 < Ksrc) ? c2 : -1;
  }
  float v = (mr >= 0 && mr < Mreal && src >= 0) ? Wsrc[(size_t)mr * Ksrc + src] : 0.0f;
  F[i] = (half_t)v;
}

// conv weights (M, C, 3, 3): fragment layout ((tap*mtn+mt)*ktn + kt)*512 + lane*16 + h
__global__ void pack_a_conv(const float* __restrict__ Wsrc, half_t* __restrict__ F,
                            int Mreal, int Ksrc, int mtn, int ktn, int kbase,
                            int kvalid, int ntot) {
  int i = blockIdx.x * blockDim.x + threadIdx.x;
  if (i >= ntot) return;
  int h = i & 15, lane = (i >> 4) & 31;
  int t = i >> 9;
  int kt = t % ktn; t /= ktn;
  int mt = t % mtn; int tap = t / mtn;
  int m = mt * 16 + (lane & 15);
  int kh = lane >> 4;
  int k = kt * 32 + a_k_of(h, kh);
  int src;
  if (kvalid == 0) src = (k < Ksrc) ? k : -1;
  else if (k < kbase) src = k;
  else {
    int s = (k - kbase) >> 4, w = (k - kbase) & 15;
    int c2 = kbase + s * kvalid + w;
    src = (w < kvalid && c2 < Ksrc) ? c2 : -1;
  }
  float v = (m < Mreal && src >= 0) ? Wsrc[((size_t)m * Ksrc + src) * 9 + tap] : 0.0f;
  F[i] = (half_t)v;
}

// ---------------------------------------------------------------------------
// GEMM (1x1 conv): C[p, m] = sum_k A[m,k] B[p, coff+k]. Branch-free.
// ---------------------------------------------------------------------------
__global__ void gemm1x1_wmma(const half_t* __restrict__ F, const float* __restrict__ B,
                             int CsB, int coff, float* __restrict__ C, int CsC, int ktn) {
  const int lane = threadIdx.x & 31;
  const int wave = threadIdx.x >> 5;
  const int p0 = (blockIdx.x * 4 + wave) * 16;
  const int mt = blockIdx.y;
  const int ln = lane & 15, kh = lane >> 4;
  const int p = p0 + ln;
  const half_t* Fp = F + ((size_t)mt * ktn) * 512 + lane * 16;
  const float* Bp0 = B + (size_t)p * CsB + coff + (kh << 4);
  v8f acc = {};
  for (int kt = 0; kt < ktn; ++kt) {
    const float* Bp = Bp0 + kt * 32;
    __builtin_prefetch(Bp + 64, 0, 1);
    const float4* q = (const float4*)Bp;
    float4 f0 = q[0], f1 = q[1], f2 = q[2], f3 = q[3];
    v16h b;
    b[0]=(half_t)f0.x; b[1]=(half_t)f0.y; b[2]=(half_t)f0.z; b[3]=(half_t)f0.w;
    b[4]=(half_t)f1.x; b[5]=(half_t)f1.y; b[6]=(half_t)f1.z; b[7]=(half_t)f1.w;
    b[8]=(half_t)f2.x; b[9]=(half_t)f2.y; b[10]=(half_t)f2.z; b[11]=(half_t)f2.w;
    b[12]=(half_t)f3.x; b[13]=(half_t)f3.y; b[14]=(half_t)f3.z; b[15]=(half_t)f3.w;
    v16h a = *(const v16h*)(Fp + (size_t)kt * 512);
    acc = __builtin_amdgcn_wmma_f32_16x16x32_f16(false, a, false, b, (short)0,
                                                 acc, false, false);
  }
  float* Cp = C + (size_t)p * CsC + mt * 16 + (kh << 3);
  float4 o0 = {acc[0], acc[1], acc[2], acc[3]};
  float4 o1 = {acc[4], acc[5], acc[6], acc[7]};
  *(float4*)Cp = o0;
  *(float4*)(Cp + 4) = o1;
}

// ---------------------------------------------------------------------------
// 3x3 SAME conv, split B source: kt<n0 -> B0, else B1. Branch-free inner loop.
// ---------------------------------------------------------------------------
__global__ void conv3x3_wmma(const half_t* __restrict__ F,
                             const float* __restrict__ B0, int Cs0, int n0,
                             const float* __restrict__ B1, int Cs1, int ktn,
                             float* __restrict__ Y, int CsY) {
  const int lane = threadIdx.x & 31;
  const int wave = threadIdx.x >> 5;
  const int p0 = (blockIdx.x * 4 + wave) * 16;
  const int mt = blockIdx.y, mtn = gridDim.y;
  const int ln = lane & 15, kh = lane >> 4;
  const int p = p0 + ln;
  const int nb = p / HW, hw = p % HW;
  const int y = hw / IMW, x = hw % IMW;
  v8f acc = {};
  for (int tap = 0; tap < 9; ++tap) {
    const int yy = y + tap / 3 - 1;
    const int xx = x + tap % 3 - 1;
    const float sc = (yy >= 0 && yy < IMW && xx >= 0 && xx < IMW) ? 1.0f : 0.0f;
    const int yyc = min(max(yy, 0), IMW - 1);
    const int xxc = min(max(xx, 0), IMW - 1);
    const size_t pr = (size_t)(nb * HW + yyc * IMW + xxc);
    const half_t* Fp = F + (((size_t)tap * mtn + mt) * ktn) * 512 + lane * 16;
    for (int kt = 0; kt < ktn; ++kt) {
      const float* Bp;
      if (kt < n0) Bp = B0 + pr * Cs0 + kt * 32 + (kh << 4);
      else         Bp = B1 + pr * Cs1 + (kt - n0) * 32 + (kh << 4);
      const float4* q = (const float4*)Bp;
      float4 f0 = q[0], f1 = q[1], f2 = q[2], f3 = q[3];
      v16h b;
      b[0]=(half_t)(sc*f0.x); b[1]=(half_t)(sc*f0.y); b[2]=(half_t)(sc*f0.z); b[3]=(half_t)(sc*f0.w);
      b[4]=(half_t)(sc*f1.x); b[5]=(half_t)(sc*f1.y); b[6]=(half_t)(sc*f1.z); b[7]=(half_t)(sc*f1.w);
      b[8]=(half_t)(sc*f2.x); b[9]=(half_t)(sc*f2.y); b[10]=(half_t)(sc*f2.z); b[11]=(half_t)(sc*f2.w);
      b[12]=(half_t)(sc*f3.x); b[13]=(half_t)(sc*f3.y); b[14]=(half_t)(sc*f3.z); b[15]=(half_t)(sc*f3.w);
      v16h a = *(const v16h*)(Fp + (size_t)kt * 512);
      acc = __builtin_amdgcn_wmma_f32_16x16x32_f16(false, a, false, b, (short)0,
                                                   acc, false, false);
    }
  }
  float* Yp = Y + (size_t)p * CsY + mt * 16 + (kh << 3);
  float4 o0 = {acc[0], acc[1], acc[2], acc[3]};
  float4 o1 = {acc[4], acc[5], acc[6], acc[7]};
  *(float4*)Yp = o0;
  *(float4*)(Yp + 4) = o1;
}

// ---------------------------------------------------------------------------
// Deformable tap GEMM. om pixel-major stride 32: [0:18) offsets, [18:27) raw
// mask. Bilinear via 4 clamped-corner b128 loads, OOB weight zeroed.
// ---------------------------------------------------------------------------
__global__ void dfconv_tap_wmma(const half_t* __restrict__ F,
                                const float* __restrict__ B0, int Cs0, int n0,
                                const float* __restrict__ B1, int Cs1, int ktn,
                                const float* __restrict__ om,
                                float* __restrict__ Y, int CsY,
                                const float* __restrict__ bias, int Mreal) {
  const int lane = threadIdx.x & 31;
  const int wave = threadIdx.x >> 5;
  const int p0 = (blockIdx.x * 4 + wave) * 16;
  const int mt = blockIdx.y, mtn = gridDim.y;
  const int ln = lane & 15, kh = lane >> 4;
  const int p = p0 + ln;
  const int nb = p / HW, hw = p % HW;
  const int y = hw / IMW, x = hw % IMW;
  const float* omp = om + (size_t)p * 32;
  v8f tot = {};
  for (int tap = 0; tap < 9; ++tap) {
    const float offy = omp[2 * tap];
    const float offx = omp[2 * tap + 1];
    const float mask = sigmoidf_(omp[18 + tap]);
    const float py = (float)(y + tap / 3 - 1) + offy;
    const float px = (float)(x + tap % 3 - 1) + offx;
    const float y0f = floorf(py), x0f = floorf(px);
    float cw[4]; size_t ca[4];
#pragma unroll
    for (int j = 0; j < 4; ++j) {
      float yi = y0f + (float)(j >> 1), xi = x0f + (float)(j & 1);
      float wt = (1.0f - fabsf(py - yi)) * (1.0f - fabsf(px - xi));
      bool val = (yi >= 0.0f) & (yi < 96.0f) & (xi >= 0.0f) & (xi < 96.0f);
      int yc = min(max((int)yi, 0), IMW - 1);
      int xc = min(max((int)xi, 0), IMW - 1);
      ca[j] = (size_t)(nb * HW + yc * IMW + xc);
      cw[j] = val ? wt : 0.0f;
    }
    v8f acc = {};
    const half_t* Fp = F + (((size_t)tap * mtn + mt) * ktn) * 512 + lane * 16;
    for (int kt = 0; kt < ktn; ++kt) {
      const float* Bb; int Cs, kloc;
      if (kt < n0) { Bb = B0; Cs = Cs0; kloc = kt * 32 + (kh << 4); }
      else         { Bb = B1; Cs = Cs1; kloc = (kt - n0) * 32 + (kh << 4); }
      v16h b;
#pragma unroll
      for (int ch = 0; ch < 4; ++ch) {
        float s0 = 0.f, s1 = 0.f, s2 = 0.f, s3 = 0.f;
#pragma unroll
        for (int j = 0; j < 4; ++j) {
          const float4 q = *(const float4*)(Bb + ca[j] * Cs + kloc + ch * 4);
          s0 += cw[j] * q.x; s1 += cw[j] * q.y; s2 += cw[j] * q.z; s3 += cw[j] * q.w;
        }
        b[ch * 4 + 0] = (half_t)s0; b[ch * 4 + 1] = (half_t)s1;
        b[ch * 4 + 2] = (half_t)s2; b[ch * 4 + 3] = (half_t)s3;
      }
      v16h a = *(const v16h*)(Fp + (size_t)kt * 512);
      acc = __builtin_amdgcn_wmma_f32_16x16x32_f16(false, a, false, b, (short)0,
                                                   acc, false, false);
    }
#pragma unroll
    for (int r = 0; r < 8; ++r) tot[r] += mask * acc[r];
  }
  float o[8];
#pragma unroll
  for (int r = 0; r < 8; ++r) {
    int m = mt * 16 + (kh << 3) + r;
    o[r] = tot[r] + ((bias && m < Mreal) ? bias[m] : 0.0f);
  }
  float* Yp = Y + (size_t)p * CsY + mt * 16 + (kh << 3);
  float4 o0 = {o[0], o[1], o[2], o[3]};
  float4 o1 = {o[4], o[5], o[6], o[7]};
  *(float4*)Yp = o0;
  *(float4*)(Yp + 4) = o1;
}

// ---------------------------------------------------------------------------
// BN stats/apply; channel c is a *real* channel index; row in buffer is
// slot-mapped when mvalid!=0 (row = (c/mvalid)*16 + c%mvalid), else row=c.
// ---------------------------------------------------------------------------
__device__ __forceinline__ int bn_row(int c, int mvalid) {
  return mvalid ? (c / mvalid) * 16 + (c % mvalid) : c;
}

__global__ void bn_stats(const float* __restrict__ X, int Cs, int mvalid,
                         float* __restrict__ mean, float* __restrict__ rstd) {
  __shared__ float ss[256], ss2[256];
  const int c = blockIdx.x;
  const int row = bn_row(c, mvalid);
  float s = 0.0f, s2 = 0.0f;
  for (int i = threadIdx.x; i < PIX; i += 256) {
    float v = X[(size_t)i * Cs + row];
    s += v; s2 += v * v;
  }
  ss[threadIdx.x] = s; ss2[threadIdx.x] = s2;
  __syncthreads();
  for (int o = 128; o > 0; o >>= 1) {
    if (threadIdx.x < o) { ss[threadIdx.x] += ss[threadIdx.x + o];
                           ss2[threadIdx.x] += ss2[threadIdx.x + o]; }
    __syncthreads();
  }
  if (threadIdx.x == 0) {
    float m = ss[0] / (float)PIX;
    float v = ss2[0] / (float)PIX - m * m;
    mean[c] = m;
    rstd[c] = rsqrtf(v + 1e-5f);
  }
}

__global__ void bn_apply(float* __restrict__ X, int Cs, int mvalid,
                         const float* __restrict__ mean, const float* __restrict__ rstd,
                         const float* __restrict__ g, const float* __restrict__ b,
                         int relu) {
  int p = blockIdx.x * blockDim.x + threadIdx.x;
  int c = blockIdx.y;
  if (p >= PIX) return;
  int row = bn_row(c, mvalid);
  float v = (X[(size_t)p * Cs + row] - mean[c]) * rstd[c] * g[c] + b[c];
  if (relu) v = fmaxf(v, 0.0f);
  X[(size_t)p * Cs + row] = v;
}

__global__ void add_bias(float* __restrict__ X, int Cs, const float* __restrict__ b) {
  int p = blockIdx.x * blockDim.x + threadIdx.x;
  int c = blockIdx.y;
  if (p >= PIX) return;
  X[(size_t)p * Cs + c] += b[c];
}

// ---------------------------------------------------------------------------
// 1-out-channel 1x1 conv over concat of two 10-ch slices (+sigmoid).
// ---------------------------------------------------------------------------
__global__ void conv1x1_to1(const float* __restrict__ A, int csA, int oA,
                            const float* __restrict__ B, int csB, int oB,
                            const float* __restrict__ w, const float* __restrict__ bb,
                            float* __restrict__ out, int so, int oo,
                            int CA, int CB, int sig) {
  int p = blockIdx.x * blockDim.x + threadIdx.x;
  if (p >= PIX) return;
  float s = bb[0];
  const float* ap = A + (size_t)p * csA + oA;
  for (int c = 0; c < CA; ++c) s += w[c] * ap[c];
  if (CB) {
    const float* bp = B + (size_t)p * csB + oB;
    for (int c = 0; c < CB; ++c) s += w[CA + c] * bp[c];
  }
  out[(size_t)p * so + oo] = sig ? sigmoidf_(s) : s;
}

// elementwise helpers (10-ch, strided) --------------------------------------
__global__ void mul_bcast(const float* __restrict__ att, int sa, int oa,
                          const float* __restrict__ X, int sx, int ox,
                          float* __restrict__ Y, int sy, int oy) {
  int p = blockIdx.x * blockDim.x + threadIdx.x;
  int c = blockIdx.y;
  if (p >= PIX) return;
  Y[(size_t)p * sy + oy + c] = att[(size_t)p * sa + oa] * X[(size_t)p * sx + ox + c];
}

__global__ void pd_finish(float* __restrict__ D, int sd, const float* __restrict__ att) {
  int p = blockIdx.x * blockDim.x + threadIdx.x;
  int c = blockIdx.y;
  if (p >= PIX) return;
  D[(size_t)p * sd + c] *= (2.0f - att[p]);
}

__global__ void accum_maps(float* __restrict__ dst, int sd, int od,
                           const float* __restrict__ src, int ss, int os) {
  int p = blockIdx.x * blockDim.x + threadIdx.x;
  int c = blockIdx.y;
  if (p >= PIX) return;
  dst[(size_t)p * sd + od + c] += src[(size_t)p * ss + os + c];
}

__global__ void avg2_maps(const float* __restrict__ A, int sa,
                          const float* __restrict__ B, int sb,
                          float* __restrict__ O, int so, int oo) {
  int p = blockIdx.x * blockDim.x + threadIdx.x;
  int c = blockIdx.y;
  if (p >= PIX) return;
  O[(size_t)p * so + oo + c] = 0.5f * (A[(size_t)p * sa + c] + B[(size_t)p * sb + c]);
}

__global__ void gate_combine(const float* __restrict__ graw, int sg,
                             const float* __restrict__ X, int sx, int ox,
                             const float* __restrict__ H2, int sh,
                             float* __restrict__ O, int so, int oo) {
  int p = blockIdx.x * blockDim.x + threadIdx.x;
  int c = blockIdx.y;
  if (p >= PIX) return;
  float g = sigmoidf_(graw[(size_t)p * sg]);
  float v = g * X[(size_t)p * sx + ox + c] + (1.0f - g) * H2[(size_t)p * sh + c];
  O[(size_t)p * so + oo + c] = fmaxf(v, 0.0f);
}

__global__ void copy_slot(float* __restrict__ dst, int csd, int od,
                          const float* __restrict__ src, int css, int os) {
  int i = blockIdx.x * blockDim.x + threadIdx.x;
  int p = i >> 4, c = i & 15;
  if (p >= PIX) return;
  dst[(size_t)p * csd + od + c] = src[(size_t)p * css + os + c];
}

__global__ void zero_buf(float* __restrict__ X, size_t n) {
  size_t i = (size_t)blockIdx.x * blockDim.x + threadIdx.x;
  if (i < n) X[i] = 0.0f;
}

__global__ void repack_pm(const float* __restrict__ in, float* __restrict__ out) {
  int i = blockIdx.x * blockDim.x + threadIdx.x;   // over PIX*256
  int p = i >> 8, c = i & 255;
  int nb = p / HW, hw = p % HW;
  out[(size_t)p * 256 + c] = in[((size_t)nb * 256 + c) * HW + hw];
}

// grouped 1x1 classifier -> NCHW ---------------------------------------------
__global__ void grouped_cls(const float* __restrict__ blk0, int cs0,
                            const float* __restrict__ blk1, int cs1,
                            const float* __restrict__ w, const float* __restrict__ b,
                            float* __restrict__ out, int G) {
  int p = blockIdx.x * blockDim.x + threadIdx.x;
  int g = blockIdx.y;
  if (p >= PIX) return;
  const float* src = (g == 0) ? (blk0 + (size_t)p * cs0)
                              : (blk1 + (size_t)p * cs1 + (g - 1) * 16);
  float s = b[g];
#pragma unroll
  for (int c = 0; c < HD_; ++c) s += w[g * HD_ + c] * src[c];
  int nb = p / HW, hw = p % HW;
  out[((size_t)nb * G + g) * HW + hw] = s;
}

__global__ void to_nchw(const float* __restrict__ in, int cs, float* __restrict__ out,
                        int C) {
  int p = blockIdx.x * blockDim.x + threadIdx.x;
  int c = blockIdx.y;
  if (p >= PIX) return;
  int nb = p / HW, hw = p % HW;
  out[((size_t)nb * C + c) * HW + hw] = in[(size_t)p * cs + c];
}

// ---------------------------------------------------------------------------
// Host-side orchestration
// ---------------------------------------------------------------------------
struct ProjP { const float *w, *g, *b; };
struct CompP { const float *w, *g, *b; };
struct DfP   { const float *w_off, *b_off, *w, *b, *bn_g, *bn_b; };
struct DecP  { const float *w, *b; };
struct PdP   { DfP d1, d2; const float *att_w, *att_b; };
struct CuP   { DfP u1, u2, gate; };
struct ClsP  { const float *w, *b; };

extern "C" void kernel_launch(void* const* d_in, const int* in_sizes, int n_in,
                              void* d_out, int out_size, void* d_ws, size_t ws_size,
                              hipStream_t stream) {
  (void)in_sizes; (void)n_in; (void)out_size; (void)ws_size;
  const int P = PIX;
  float* Wb = (float*)d_ws;

  // ---- activation arena (pixel-major, bump-allocated; deterministic) ----
  size_t actCur = 0;
  auto abuf = [&](int Cs) { float* r = Wb + actCur; actCur += (size_t)Cs * P; return r; };
  float* FEA_P = abuf(256);  float* FEA_H = abuf(256);  float* FEA_F = abuf(256);
  float* XP_C  = abuf(112);  float* XH_C  = abuf(48);   float* XF_C  = abuf(32);
  float* COMP_H = abuf(16);  float* COMP_P = abuf(16);
  float* COMP_U = abuf(16);  float* COMP_L = abuf(16);
  float* ATTXF_U = abuf(16); float* ATTXF_L = abuf(16);
  float* DP_U = abuf(16);    float* DP_L = abuf(16);
  float* ATT_ALL = abuf(8);  float* ATT_FP = abuf(8);   float* ATT_HP = abuf(8);
  float* DEC_FP = abuf(96);  float* DEC_HP = abuf(96);
  float* XPP = abuf(96);     float* XP_NEW = abuf(96);
  float* XH_UL = abuf(32);   float* XF_NEW = abuf(16);
  float* CC = abuf(64);      float* D1CC = abuf(32);    float* D1 = abuf(32);
  float* H1 = abuf(32);      float* H2 = abuf(16);      float* GATE = abuf(16);
  float* OM = abuf(32);      float* PDS = abuf(16);     float* ATT1 = abuf(16);
  const size_t actTotal = actCur;

  // ---- weight-fragment arena + stats ----
  half_t* FRbase = (half_t*)(Wb + actTotal);
  size_t frCur = 0;
  auto fragAlloc = [&](size_t nh) { half_t* r = FRbase + frCur; frCur += nh; return r; };
  float* stat_mean = (float*)(FRbase + (size_t)12 * 1024 * 1024);
  float* stat_rstd = stat_mean + 128;

  // ---- parameter extraction (setup_inputs dict insertion order) ----
  int cur = 3;
  auto nxt = [&]() { return (const float*)d_in[cur++]; };
  auto get_proj = [&]() { ProjP q; q.w = nxt(); q.g = nxt(); q.b = nxt(); return q; };
  auto get_comp = [&]() { CompP q; q.w = nxt(); q.g = nxt(); q.b = nxt(); return q; };
  auto get_df = [&](bool bias, bool bn) {
    DfP q{}; q.w_off = nxt(); q.b_off = nxt(); q.w = nxt();
    if (bias) q.b = nxt();
    if (bn) { q.bn_g = nxt(); q.bn_b = nxt(); }
    return q;
  };
  auto get_dec = [&]() { DecP q; q.w = nxt(); q.b = nxt(); return q; };
  auto get_pd = [&]() {
    PdP q; q.d1 = get_df(false, true); q.d2 = get_df(false, true);
    q.att_w = nxt(); q.att_b = nxt(); return q;
  };
  auto get_cu = [&]() {
    CuP q; q.u1 = get_df(false, true); q.u2 = get_df(false, true);
    q.gate = get_df(true, false); return q;
  };
  auto get_cls = [&]() { ClsP q; q.w = nxt(); q.b = nxt(); return q; };

  ProjP p_conv = get_proj(), h_conv = get_proj(), f_conv = get_proj();
  CompP comp_h_p = get_comp(), comp_p_p = get_comp();
  CuP   cu_full = get_cu();
  CompP comp_phu = get_comp(), comp_phl = get_comp();
  DecP  decomp_fhu = get_dec(), decomp_fhl = get_dec();
  PdP   dp_u_p = get_pd(), dp_l_p = get_pd();
  CuP   update_u = get_cu();
  PdP   dp_e[10]; for (int i = 0; i < 10; ++i) dp_e[i] = get_pd();
  DecP  dec_fp[6]; for (int i = 0; i < 6; ++i) dec_fp[i] = get_dec();
  DecP  dec_hp[6]; for (int i = 0; i < 6; ++i) dec_hp[i] = get_dec();
  CuP   cu_part[6]; for (int i = 0; i < 6; ++i) cu_part[i] = get_cu();
  ClsP  pg = get_cls(), hg = get_cls(), fg = get_cls();

  // ---- launch helpers ----
  auto packG = [&](const float* Ws, int Mreal, int Ksrc, int mtn, int ktn,
                   int kbase, int kvalid, int mvalid) -> const half_t* {
    size_t n = (size_t)mtn * ktn * 512;
    half_t* F = fragAlloc(n);
    pack_a_gemm<<<dim3((unsigned)((n + 255) / 256)), 256, 0, stream>>>(
        Ws, F, Mreal, Ksrc, ktn, kbase, kvalid, mvalid, (int)n);
    return F;
  };
  auto packC = [&](const float* Ws, int Mreal, int Ksrc, int mtn, int ktn,
                   int kbase, int kvalid) -> const half_t* {
    size_t n = (size_t)9 * mtn * ktn * 512;
    half_t* F = fragAlloc(n);
    pack_a_conv<<<dim3((unsigned)((n + 255) / 256)), 256, 0, stream>>>(
        Ws, F, Mreal, Ksrc, mtn, ktn, kbase, kvalid, (int)n);
    return F;
  };
  auto gemm = [&](const half_t* F, const float* B, int CsB, int coff,
                  float* C, int CsC, int mtn, int ktn) {
    gemm1x1_wmma<<<dim3(P / 64, mtn), 128, 0, stream>>>(F, B, CsB, coff, C, CsC, ktn);
  };
  auto bnact = [&](float* X, int Cs, int mvalid, const float* g, const float* b,
                   int Creal, int relu) {
    bn_stats<<<Creal, 256, 0, stream>>>(X, Cs, mvalid, stat_mean, stat_rstd);
    bn_apply<<<dim3(P / 256, Creal), 256, 0, stream>>>(X, Cs, mvalid, stat_mean,
                                                       stat_rstd, g, b, relu);
  };
  auto cps = [&](float* dst, int csd, int od, const float* src, int css, int os) {
    copy_slot<<<P * 16 / 256, 256, 0, stream>>>(dst, csd, od, src, css, os);
  };
  // dfconv over input described by split-B (B0 for kt<n0, else B1)
  auto dfconv = [&](const float* B0, int Cs0, int n0, const float* B1, int Cs1,
                    int ktn, int cinReal, int kbase, int kvalid,
                    int cout, const DfP& q, float* out, int CsOut, bool bnrelu) {
    int mtn_o = (27 + 15) / 16;
    int mtn_w = (cout + 15) / 16;
    const half_t* Fo = packC(q.w_off, 27, cinReal, mtn_o, ktn, kbase, kvalid);
    const half_t* Fw = packC(q.w, cout, cinReal, mtn_w, ktn, kbase, kvalid);
    conv3x3_wmma<<<dim3(P / 64, mtn_o), 128, 0, stream>>>(Fo, B0, Cs0, n0, B1, Cs1,
                                                          ktn, OM, 32);
    add_bias<<<dim3(P / 256, 27), 256, 0, stream>>>(OM, 32, q.b_off);
    dfconv_tap_wmma<<<dim3(P / 64, mtn_w), 128, 0, stream>>>(
        Fw, B0, Cs0, n0, B1, Cs1, ktn, OM, out, CsOut, q.b, cout);
    if (bnrelu) bnact(out, CsOut, 0, q.bn_g, q.bn_b, cout, 1);
  };
  auto composition = [&](const float* src, int csS, int coff, int np,
                         const CompP& c, float* out) {
    const half_t* F = packG(c.w, HD_, np * HD_, 1, np / 2, 0, HD_, 0);
    gemm(F, src, csS, coff, out, 16, 1, np / 2);
    bnact(out, 16, 0, c.g, c.b, HD_, 1);
  };
  auto decomposition = [&](const float* A, int csA, int oA, const float* B,
                           int csB, int oB, const DecP& d, float* att, int so, int oo) {
    conv1x1_to1<<<P / 256, 256, 0, stream>>>(A, csA, oA, B, csB, oB, d.w, d.b,
                                             att, so, oo, HD_, HD_, 1);
  };
  auto part_dep = [&](const float* pA, int csA, int oA,
                      const float* pB, int csB, int oB, const PdP& q, float* out) {
    cps(D1CC, 32, 0, pA, csA, oA);
    cps(D1CC, 32, 16, pB, csB, oB);
    dfconv(D1CC, 32, 1, D1CC, 32, 1, 20, 0, HD_, 20, q.d1, D1, 32, true);
    dfconv(D1, 32, 1, D1, 32, 1, 20, 0, 0, HD_, q.d2, out, 16, true);
    conv1x1_to1<<<P / 256, 256, 0, stream>>>(pA, csA, oA, pA, csA, oA, q.att_w,
                                             q.att_b, ATT1, 16, 0, HD_, 0, 1);
    pd_finish<<<dim3(P / 256, HD_), 256, 0, stream>>>(out, 16, ATT1);
  };
  auto conv_update = [&](const float* fea, const float* xs, int xcs, int xo,
                         const float* m1, int c1, int o1,
                         const float* m2, int c2, int o2,
                         const float* m3, int c3, int o3,
                         const CuP& q, float* out, int so, int oo) {
    cps(CC, 64, 0, xs, xcs, xo);
    cps(CC, 64, 16, m1, c1, o1);
    cps(CC, 64, 32, m2, c2, o2);
    int cinReal = 286;
    if (m3) { cps(CC, 64, 48, m3, c3, o3); cinReal = 296; }
    else    { cps(CC, 64, 48, ATT1, 16, 0); /* overwritten-by-zero region; A cols are zero */ }
    dfconv(fea, 256, 8, CC, 64, 10, cinReal, 256, HD_, 20, q.u1, H1, 32, true);
    dfconv(H1, 32, 1, H1, 32, 1, 20, 0, 0, 1, q.gate, GATE, 16, false);
    dfconv(H1, 32, 1, H1, 32, 1, 20, 0, 0, HD_, q.u2, H2, 16, true);
    gate_combine<<<dim3(P / 256, HD_), 256, 0, stream>>>(GATE, 16, xs, xcs, xo,
                                                         H2, 16, out, so, oo);
  };

  // ---- stage 0: zero activation arena (pad hygiene), repack inputs ----
  zero_buf<<<(unsigned)((actTotal + 255) / 256), 256, 0, stream>>>(Wb, actTotal);
  repack_pm<<<P * 256 / 256, 256, 0, stream>>>((const float*)d_in[0], FEA_P);
  repack_pm<<<P * 256 / 256, 256, 0, stream>>>((const float*)d_in[1], FEA_H);
  repack_pm<<<P * 256 / 256, 256, 0, stream>>>((const float*)d_in[2], FEA_F);

  // ---- projections (slot-mapped output rows) + BN + ReLU ----
  gemm(packG(p_conv.w, 70, 256, 7, 8, 0, 0, HD_), FEA_P, 256, 0, XP_C, 112, 7, 8);
  bnact(XP_C, 112, HD_, p_conv.g, p_conv.b, 70, 1);
  gemm(packG(h_conv.w, 30, 256, 3, 8, 0, 0, HD_), FEA_H, 256, 0, XH_C, 48, 3, 8);
  bnact(XH_C, 48, HD_, h_conv.g, h_conv.b, 30, 1);
  gemm(packG(f_conv.w, 20, 256, 2, 8, 0, 0, HD_), FEA_F, 256, 0, XF_C, 32, 2, 8);
  bnact(XF_C, 32, HD_, f_conv.g, f_conv.b, 20, 1);

  // slot offsets: group i lives at channel 16*i of its buffer
  // xf_n = XF_C slot1; xh_n0 = XH_C slot1; xh_n1 = XH_C slot2; xp_n(i) = XP_C slot(1+i)

  // ---- full branch ----
  composition(XH_C, 48, 16, 2, comp_h_p, COMP_H);
  composition(XP_C, 112, 16, 6, comp_p_p, COMP_P);
  conv_update(FEA_F, XF_C, 32, 16, COMP_H, 16, 0, COMP_P, 16, 0,
              nullptr, 0, 0, cu_full, XF_NEW, 16, 0);

  // ---- half branch ----
  composition(XP_C, 112, 16, 4, comp_phu, COMP_U);
  part_dep(XH_C, 48, 32, XH_C, 48, 16, dp_u_p, DP_U);
  decomposition(XF_C, 32, 16, XH_C, 48, 16, decomp_fhu, ATT_ALL, 8, 0);     // att_u
  mul_bcast<<<dim3(P / 256, HD_), 256, 0, stream>>>(ATT_ALL, 8, 0, XF_C, 32, 16,
                                                    ATTXF_U, 16, 0);
  conv_update(FEA_H, XH_C, 48, 16, COMP_U, 16, 0, ATTXF_U, 16, 0, DP_U, 16, 0,
              update_u, XH_UL, 32, 0);

  composition(XP_C, 112, 80, 2, comp_phl, COMP_L);
  part_dep(XH_C, 48, 16, XH_C, 48, 32, dp_l_p, DP_L);
  decomposition(XF_C, 32, 16, XH_C, 48, 32, decomp_fhl, ATT_ALL, 8, 1);     // att_l
  mul_bcast<<<dim3(P / 256, HD_), 256, 0, stream>>>(ATT_ALL, 8, 1, XF_C, 32, 16,
                                                    ATTXF_L, 16, 0);
  conv_update(FEA_H, XH_C, 48, 32, COMP_L, 16, 0, ATTXF_L, 16, 0, DP_L, 16, 0,
              update_u, XH_UL, 32, 16);

  // ---- part branch: decompositions ----
  for (int i = 0; i < 6; ++i) {
    int xpo = 16 * (1 + i);
    decomposition(XF_C, 32, 16, XP_C, 112, xpo, dec_fp[i], ATT_FP, 8, i);
    mul_bcast<<<dim3(P / 256, HD_), 256, 0, stream>>>(ATT_FP, 8, i, XF_C, 32, 16,
                                                      DEC_FP, 96, 16 * i);
    int xho = (i < 4) ? 16 : 32;                                            // UPPER
    decomposition(XH_C, 48, xho, XP_C, 112, xpo, dec_hp[i], ATT_HP, 8, i);
    mul_bcast<<<dim3(P / 256, HD_), 256, 0, stream>>>(ATT_HP, 8, i, XH_C, 48, xho,
                                                      DEC_HP, 96, 16 * i);
  }

  // ---- part messages over graph edges ----
  static const int EA[10] = {0, 1, 1, 1, 2, 2, 3, 4, 4, 5};
  static const int EB[10] = {1, 0, 2, 4, 1, 3, 2, 1, 5, 4};
  for (int e = 0; e < 10; ++e) {
    part_dep(XP_C, 112, 16 * (1 + EA[e]), XP_C, 112, 16 * (1 + EB[e]), dp_e[e], PDS);
    accum_maps<<<dim3(P / 256, HD_), 256, 0, stream>>>(XPP, 96, 16 * EB[e], PDS, 16, 0);
  }

  // ---- part updates ----
  for (int j = 0; j < 6; ++j) {
    conv_update(FEA_P, XP_C, 112, 16 * (1 + j),
                DEC_FP, 96, 16 * j, DEC_HP, 96, 16 * j, XPP, 96, 16 * j,
                cu_part[j], XP_NEW, 96, 16 * j);
  }

  // ---- attention maps: att_p = 0.5*(att_fp + att_hp) ----
  avg2_maps<<<dim3(P / 256, 6), 256, 0, stream>>>(ATT_FP, 8, ATT_HP, 8, ATT_ALL, 8, 2);

  // ---- heads ----
  float* out = (float*)d_out;
  float* out_xp = out;                         // 2*7*HW
  float* out_xh = out + (size_t)2 * 7 * HW;    // 2*3*HW
  float* out_xf = out_xh + (size_t)2 * 3 * HW; // 2*2*HW
  float* out_at = out_xf + (size_t)2 * 2 * HW; // 2*8*HW
  grouped_cls<<<dim3(P / 256, 7), 256, 0, stream>>>(XP_C, 112, XP_NEW, 96,
                                                    pg.w, pg.b, out_xp, 7);
  grouped_cls<<<dim3(P / 256, 3), 256, 0, stream>>>(XH_C, 48, XH_UL, 32,
                                                    hg.w, hg.b, out_xh, 3);
  grouped_cls<<<dim3(P / 256, 2), 256, 0, stream>>>(XF_C, 32, XF_NEW, 16,
                                                    fg.w, fg.b, out_xf, 2);
  to_nchw<<<dim3(P / 256, 8), 256, 0, stream>>>(ATT_ALL, 8, out_at, 8);
}
